// GammaFFNNLayer_60790967107630
// MI455X (gfx1250) — matrix operand order repaired and checked by reading, not compiled, not run on hardware
//
#include <hip/hip_runtime.h>

typedef __attribute__((ext_vector_type(16))) _Float16 v16h;
typedef __attribute__((ext_vector_type(8)))  float    v8f;

#define LOG2E 1.4426950408889634f
#define LN2   0.6931471805599453f

__device__ __forceinline__ float fast_tanh(float x) {
#if __has_builtin(__builtin_amdgcn_tanhf)
    return __builtin_amdgcn_tanhf(x);          // native v_tanh_f32 (TRANS pipe)
#else
    // tanh(x) = 1 - 2/(exp(2x)+1); exact limits at +/-inf
    float e = __builtin_amdgcn_exp2f(x * (2.0f * LOG2E));
    return 1.0f - 2.0f * __builtin_amdgcn_rcpf(e + 1.0f);
#endif
}

__device__ __forceinline__ float softplus(float x) {
    // log1p(exp(x)) = max(x,0) + log(1 + exp(-|x|))
    float e = __builtin_amdgcn_exp2f(-fabsf(x) * LOG2E);
    return fmaxf(x, 0.0f) + LN2 * __builtin_amdgcn_logf(1.0f + e);
}

// One MLP evaluation f(x, g) for 16 rows batched across the wave.
// Row r = lane&15 (duplicated across wave halves); returns f(row) on every lane.
__device__ __forceinline__ float mlp_f(
    float x, float g,
    const float (&w1x)[16], const float (&w1g)[16], const float (&bb1)[16],
    v16h B0, v16h B1,
    float b2c0, float b2c1, float w3c0, float w3c1, float b3s,
    int lane)
{
    // ---- layer 1 (2 -> 32) computed directly in WMMA A-operand K-order ----
    v16h A;
#pragma unroll
    for (int i = 0; i < 16; ++i) {
        float h = fast_tanh(fmaf(x, w1x[i], fmaf(g, w1g[i], bb1[i])));
        A[i] = (_Float16)h;
    }

    // ---- layer 2 (32 -> 32) on the matrix pipes: two 16x16x32 WMMAs ----
    v8f c0 = {0.f, 0.f, 0.f, 0.f, 0.f, 0.f, 0.f, 0.f};
    v8f d0 = __builtin_amdgcn_wmma_f32_16x16x32_f16(false, A, false, B0, (short)0, c0, false, false);
    v8f d1 = __builtin_amdgcn_wmma_f32_16x16x32_f16(false, A, false, B1, (short)0, c0, false, false);

    // ---- bias + tanh + layer-3 partials (this lane owns 2 of 32 columns) ----
    // a[v] = partial for value-index v (row = v + 8*half of the lane holding it)
    float a[8];
#pragma unroll
    for (int v = 0; v < 8; ++v) {
        float p0 = fast_tanh(d0[v] + b2c0) * w3c0;
        float p1 = fast_tanh(d1[v] + b2c1) * w3c1;
        a[v] = p0 + p1;
    }

    // ---- routing reduction across the 16 lanes of each half: 9 shuffles total ----
    const bool p0b = (lane & 1) != 0;
    const bool p1b = (lane & 2) != 0;
    const bool p2b = (lane & 4) != 0;

    // step 0 (xor 1): 8 -> 4 values; keep v with bit0(v)==bit0(lane)
    float b[4];
#pragma unroll
    for (int j = 0; j < 4; ++j) {
        float kept = p0b ? a[(j << 1) | 1] : a[(j << 1) | 0];
        float give = p0b ? a[(j << 1) | 0] : a[(j << 1) | 1];
        b[j] = kept + __shfl_xor(give, 1, 32);
    }
    // step 1 (xor 2): 4 -> 2 values; keep bit1(v)==bit1(lane)
    float c[2];
#pragma unroll
    for (int i = 0; i < 2; ++i) {
        float kept = p1b ? b[(i << 1) | 1] : b[(i << 1) | 0];
        float give = p1b ? b[(i << 1) | 0] : b[(i << 1) | 1];
        c[i] = kept + __shfl_xor(give, 2, 32);
    }
    // step 2 (xor 4): 2 -> 1 value; keep bit2(v)==bit2(lane)
    float kept2 = p2b ? c[1] : c[0];
    float give2 = p2b ? c[0] : c[1];
    float d = kept2 + __shfl_xor(give2, 4, 32);
    // step 3 (xor 8): finish the 16-lane half-reduction
    d += __shfl_xor(d, 8, 32);
    // lane q now holds total[row = (q&7) + 8*half]; fix the cross-half cases
    float other = __shfl_xor(d, 16, 32);
    bool useOwn = (((lane >> 4) & 1) == ((lane >> 3) & 1));
    float pre = useOwn ? d : other;

    return softplus(pre + b3s);
}

#define WAVES_PER_BLOCK 8

__global__ __launch_bounds__(256) void rk4_gamma_kernel(
    const float* __restrict__ inputs, const float* __restrict__ gamma,
    const float* __restrict__ W1, const float* __restrict__ b1,
    const float* __restrict__ W2, const float* __restrict__ b2,
    const float* __restrict__ W3, const float* __restrict__ b3,
    float* __restrict__ out, int n, int tiles_total, int waves_total)
{
    const int lane = threadIdx.x & 31;
    const int wave = threadIdx.x >> 5;
    const int wave_global = blockIdx.x * WAVES_PER_BLOCK + wave;
    const int half = lane >> 4;
    const int col  = lane & 15;

    // ---- per-lane weight caches (loaded once per wave) ----
    // Layer-1 weights in the A-operand K-order for this lane's half.
    float w1x[16], w1g[16], bb1[16];
#pragma unroll
    for (int i = 0; i < 16; ++i) {
        int k = (i & 7) + (half << 3) + ((i >> 3) << 4);
        w1x[i] = W1[k];        // W1[0][k], W1 is (2,32) row-major
        w1g[i] = W1[32 + k];   // W1[1][k]
        bb1[i] = b1[k];
    }
    // W2 (32x32, row-major) as two B operands (cols 0..15 and 16..31),
    // B-operand K-order: element i holds K = (i&7) + half*8 + (i>>3)*16 for column `col`.
    v16h B0, B1;
#pragma unroll
    for (int i = 0; i < 16; ++i) {
        int k = (i & 7) + (half << 3) + ((i >> 3) << 4);
        B0[i] = (_Float16)W2[k * 32 + col];
        B1[i] = (_Float16)W2[k * 32 + col + 16];
    }
    const float b2c0 = b2[col], b2c1 = b2[col + 16];
    const float w3c0 = W3[col], w3c1 = W3[col + 16];  // W3 is (32,1)
    const float b3s  = b3[0];

    // ---- grid-stride loop over 16-row tiles ----
    for (int t = wave_global; t < tiles_total; t += waves_total) {
        int row  = t * 16 + col;
        int rowc = row < n ? row : (n - 1);   // arithmetic clamp keeps EXEC full

        const float4 inp = *(const float4*)(inputs + (size_t)rowc * 4);
        const float g0 = gamma[rowc];
        const float eps_n = inp.x, eps_h = inp.y, eps_o = inp.z, dt = inp.w;

        float f1 = mlp_f(eps_n, g0, w1x, w1g, bb1, B0, B1, b2c0, b2c1, w3c0, w3c1, b3s, lane);
        float k1 = dt * f1 * (eps_n - g0);
        float g1 = fmaf(0.5f, k1, g0);

        float f2 = mlp_f(eps_h, g1, w1x, w1g, bb1, B0, B1, b2c0, b2c1, w3c0, w3c1, b3s, lane);
        float k2 = dt * f2 * (eps_h - g1);
        float g2 = fmaf(0.5f, k2, g0);

        float f3 = mlp_f(eps_h, g2, w1x, w1g, bb1, B0, B1, b2c0, b2c1, w3c0, w3c1, b3s, lane);
        float k3 = dt * f3 * (eps_h - g2);
        float g3 = g0 + k3;

        float f4 = mlp_f(eps_o, g3, w1x, w1g, bb1, B0, B1, b2c0, b2c1, w3c0, w3c1, b3s, lane);
        float k4 = dt * f4 * (eps_o - g3);

        float gnew = g0 + (k1 + 2.0f * k2 + 2.0f * k3 + k4) * (1.0f / 6.0f);

        if (half == 0 && row < n) out[row] = gnew;   // divergence only after all WMMAs
    }
}

extern "C" void kernel_launch(void* const* d_in, const int* in_sizes, int n_in,
                              void* d_out, int out_size, void* d_ws, size_t ws_size,
                              hipStream_t stream) {
    const float* inputs = (const float*)d_in[0];
    const float* gamma  = (const float*)d_in[1];
    const float* W1     = (const float*)d_in[2];
    const float* b1     = (const float*)d_in[3];
    const float* W2     = (const float*)d_in[4];
    const float* b2     = (const float*)d_in[5];
    const float* W3     = (const float*)d_in[6];
    const float* b3     = (const float*)d_in[7];
    float* out = (float*)d_out;

    int n = in_sizes[0] / 4;                 // inputs is (B, 4)
    int tiles = (n + 15) / 16;
    int blocks = 2048;                       // 16384 waves; 4 tiles/wave at B=2^20
    int waves_total = blocks * WAVES_PER_BLOCK;
    if (waves_total > tiles) {
        blocks = (tiles + WAVES_PER_BLOCK - 1) / WAVES_PER_BLOCK;
        if (blocks < 1) blocks = 1;
        waves_total = blocks * WAVES_PER_BLOCK;
    }

    rk4_gamma_kernel<<<blocks, 256, 0, stream>>>(
        inputs, gamma, W1, b1, W2, b2, W3, b3, out, n, tiles, waves_total);
}
